// Net_85152021610649
// MI455X (gfx1250) — matrix operand order repaired and checked by reading, not compiled, not run on hardware
//
#include <hip/hip_runtime.h>

typedef __attribute__((ext_vector_type(16))) _Float16 v16h;
typedef __attribute__((ext_vector_type(8)))  float    v8f;

#define T_SEQ 50
#define N_IN  20
#define N_HID 128
#define N_CAT 10
#define ROWS  16       // batch rows per block
#define HS    67       // dword stride of one h row in LDS (64 data + 3 pad) -> conflict-free A reads
#define PS    129      // f32 stride of pooled row

__device__ __forceinline__ float fsigmoid(float x) {
    return 1.0f / (1.0f + __expf(-x));
}
__device__ __forceinline__ float ftanh_fast(float x) {
    return 2.0f / (1.0f + __expf(-2.0f * x)) - 1.0f;
}

__global__ __launch_bounds__(256)
void lstm_wmma_kernel(const float* __restrict__ x,
                      const float* __restrict__ W_ih,
                      const float* __restrict__ W_hh,
                      const float* __restrict__ b_ih,
                      const float* __restrict__ b_hh,
                      const float* __restrict__ W_fc,
                      const float* __restrict__ b_fc,
                      float* __restrict__ out)
{
    __shared__ unsigned int hbuf[2][ROWS * HS];   // double-buffered h(t), packed f16 pairs
    __shared__ float pooled[ROWS * PS];           // temporal-max h, f32

    const int tid  = threadIdx.x;
    const int wv   = tid >> 5;      // wave 0..7 : owns gate cols wv*16..wv*16+15 of each gate block
    const int lane = tid & 31;
    const int lm   = lane & 15;     // A: batch row within tile; B/C: N column within tile
    const int hi   = lane >> 4;
    const int bRow = blockIdx.x * ROWS;

    // zero h(0) (both buffers)
    for (int i = tid; i < 2 * ROWS * HS; i += 256) ((unsigned int*)hbuf)[i] = 0u;

    // ---- bias (i,f,g,o blocks), broadcast into accumulators each step
    float bias[4];
#pragma unroll
    for (int nt = 0; nt < 4; ++nt) {
        int n = nt * N_HID + wv * 16 + lm;
        bias[nt] = b_ih[n] + b_hh[n];
    }

    // ---- stationary B fragments in VGPRs.
    // B 32x16 f16 frag, this code's contiguous K order (same as A): lane -> N = lm; VGPR v holds K = 2v + hi*16 (pair).
    // Bf[nt][kt<4] = W_hh K-tile kt; Bf[nt][4] = W_ih padded to K=32.
    v16h Bf[4][5];
#pragma unroll
    for (int nt = 0; nt < 4; ++nt) {
        int n = nt * N_HID + wv * 16 + lm;
        const float* wr = W_hh + (size_t)n * N_HID;
#pragma unroll
        for (int kt = 0; kt < 4; ++kt) {
#pragma unroll
            for (int v = 0; v < 8; ++v) {
                int k = kt * 32 + 2 * v + hi * 16;
                float2 p = *(const float2*)(wr + k);
                Bf[nt][kt][2 * v]     = (_Float16)p.x;
                Bf[nt][kt][2 * v + 1] = (_Float16)p.y;
            }
        }
        const float* wx = W_ih + (size_t)n * N_IN;
#pragma unroll
        for (int v = 0; v < 8; ++v) {
            int k = 2 * v + hi * 16;
            float2 p = make_float2(0.f, 0.f);
            if (k < N_IN) p = *(const float2*)(wx + k);   // k even, so pair fully valid
            Bf[nt][4][2 * v]     = (_Float16)p.x;
            Bf[nt][4][2 * v + 1] = (_Float16)p.y;
        }
    }

    // cell state + running max of h, per C/D element (row = r + 8*hi, col = wv*16 + lm)
    float cst[8], hmax[8];
#pragma unroll
    for (int r = 0; r < 8; ++r) { cst[r] = 0.f; hmax[r] = -3.0e38f; }

    const float* xrow = x + (size_t)(bRow + lm) * T_SEQ * N_IN;

    __syncthreads();

    for (int t = 0; t < T_SEQ; ++t) {
        v8f acc[4];
#pragma unroll
        for (int nt = 0; nt < 4; ++nt)
#pragma unroll
            for (int r = 0; r < 8; ++r) acc[nt][r] = bias[nt];

        // ---- x_t contribution: A frag (16x32 f16, K padded from 20) from global
        v16h ax;
#pragma unroll
        for (int v = 0; v < 8; ++v) {
            int k = 2 * v + (hi << 4);  // A-layout K index
            float2 p = make_float2(0.f, 0.f);
            if (k < N_IN) p = *(const float2*)(xrow + t * N_IN + k);
            ax[2 * v]     = (_Float16)p.x;
            ax[2 * v + 1] = (_Float16)p.y;
        }
#pragma unroll
        for (int nt = 0; nt < 4; ++nt)
            acc[nt] = __builtin_amdgcn_wmma_f32_16x16x32_f16(
                false, ax, false, Bf[nt][4], (short)0, acc[nt], false, false);

        // ---- h(t) contribution: 4 K-tiles, A frags from LDS (b32 = f16 pair)
        const unsigned int* hrd = hbuf[t & 1] + lm * HS;
#pragma unroll
        for (int kt = 0; kt < 4; ++kt) {
            v16h ah;
#pragma unroll
            for (int v = 0; v < 8; ++v) {
                int k = kt * 32 + 2 * v + (hi << 4);
                union { unsigned int u; _Float16 h[2]; } cv;
                cv.u = hrd[k >> 1];
                ah[2 * v]     = cv.h[0];
                ah[2 * v + 1] = cv.h[1];
            }
#pragma unroll
            for (int nt = 0; nt < 4; ++nt)
                acc[nt] = __builtin_amdgcn_wmma_f32_16x16x32_f16(
                    false, ah, false, Bf[nt][kt], (short)0, acc[nt], false, false);
        }

        // ---- LSTM cell elementwise (wave holds i,f,g,o for its own hidden cols)
        unsigned int* hwr = hbuf[(t + 1) & 1];
#pragma unroll
        for (int r = 0; r < 8; ++r) {
            float ig = fsigmoid(acc[0][r]);
            float fg = fsigmoid(acc[1][r]);
            float gg = ftanh_fast(acc[2][r]);
            float og = fsigmoid(acc[3][r]);
            float cn = fg * cst[r] + ig * gg;
            cst[r] = cn;
            float h = og * ftanh_fast(cn);
            hmax[r] = fmaxf(hmax[r], h);
            // pack col-pairs into b32 via shuffle, even-col lanes store (conflict-free)
            float hN = __shfl_xor(h, 1, 32);
            if ((lm & 1) == 0) {
                union { unsigned int u; _Float16 h[2]; } cv;
                cv.h[0] = (_Float16)h;
                cv.h[1] = (_Float16)hN;
                int m = r + (hi << 3);
                hwr[m * HS + (wv << 3) + (lm >> 1)] = cv.u;
            }
        }
        __syncthreads();
    }

    // ---- stage temporal-max pooled h to LDS
#pragma unroll
    for (int r = 0; r < 8; ++r) {
        int m = r + (hi << 3);
        pooled[m * PS + (wv << 4) + lm] = hmax[r];
    }
    __syncthreads();

    // ---- FC (128 -> 10) + log_softmax, one lane per batch row (tiny)
    if (tid < ROWS) {
        const float* pr = pooled + tid * PS;
        float logits[N_CAT];
        float mx = -3.0e38f;
#pragma unroll
        for (int c = 0; c < N_CAT; ++c) {
            const float* wf = W_fc + c * N_HID;
            float s = b_fc[c];
            for (int j = 0; j < N_HID; ++j) s += pr[j] * wf[j];
            logits[c] = s;
            mx = fmaxf(mx, s);
        }
        float sum = 0.f;
#pragma unroll
        for (int c = 0; c < N_CAT; ++c) sum += __expf(logits[c] - mx);
        float lse = mx + __logf(sum);
        float* orow = out + (size_t)(bRow + tid) * N_CAT;
#pragma unroll
        for (int c = 0; c < N_CAT; ++c) orow[c] = logits[c] - lse;
    }
}

extern "C" void kernel_launch(void* const* d_in, const int* in_sizes, int n_in,
                              void* d_out, int out_size, void* d_ws, size_t ws_size,
                              hipStream_t stream) {
    const float* x    = (const float*)d_in[0];
    const float* W_ih = (const float*)d_in[1];
    const float* W_hh = (const float*)d_in[2];
    const float* b_ih = (const float*)d_in[3];
    const float* b_hh = (const float*)d_in[4];
    const float* W_fc = (const float*)d_in[5];
    const float* b_fc = (const float*)d_in[6];
    float* out = (float*)d_out;

    const int batch  = in_sizes[0] / (T_SEQ * N_IN);   // 4096
    const int blocks = batch / ROWS;                   // 256

    lstm_wmma_kernel<<<blocks, 256, 0, stream>>>(x, W_ih, W_hh, b_ih, b_hh, W_fc, b_fc, out);
}
